// Net_ht_59072980189450
// MI455X (gfx1250) — compile-verified, matrix operand order checked by reading, not run on hardware
//
#include <hip/hip_runtime.h>

// ---------------------------------------------------------------------------
// Hierarchical-Tucker network, restructured as 3 GEMMs per block + tiny preps.
//   P1: T1[(z,q,r),(d,e)] = X[z,p,q,r] . U1[p,(d,e)]          (K = p <= 16)
//   P2: T2[(z,r,d),(f,h)] = T1 . C,  C[(q,e),(f,h)] = sum_g B2[e,g,f] U2[q,h,g]
//   P3: Y[(z,d,h),m] = relu( T2 . E ),  E[(f,r),m] = sum_k B1[f,k] U3[r,m,k]
// All GEMMs run on v_wmma_f32_16x16x32_bf16 (f32 accumulate; bf16 keeps f32
// exponent range -- activations grow ~1000x per block, f16 would overflow).
// Block shapes are compile-time template params (3 instantiations) so all
// div/mod in the permuted-store epilogues folds to shifts/magic muls, and all
// indexing is 32-bit to avoid v_mul_u64 address chains.
// ---------------------------------------------------------------------------

typedef __bf16 bhalf;
typedef __attribute__((ext_vector_type(16))) __bf16 v16bf;
typedef __attribute__((ext_vector_type(8)))  __bf16 v8bf;
typedef __attribute__((ext_vector_type(8)))  float  v8f;

static __device__ __forceinline__ v8f wmma_bf16(v16bf a, v16bf b, v8f c) {
  // (neg_a, A, neg_b, B, c_mod, C, reuse_a, reuse_b)
  return __builtin_amdgcn_wmma_f32_16x16x32_bf16(false, a, false, b, (short)0, c,
                                                 false, false);
}

// A/B fragment element i of lane (half = lane>>4) holds K index:
static __device__ __forceinline__ int frag_k(int half, int i) {
  return 8 * half + i + ((i >= 8) ? 8 : 0);
}

// Load one 16(M) x 32(K) A tile from row-major bf16, row length LDA (= K).
// K is a multiple of 16, so 8-element chunks are all-in or all-out; guards
// constant-fold when k0 is a constant and K % 32 == 0.
template <int LDA, int K>
static __device__ __forceinline__ v16bf load_a_tile(const bhalf* __restrict__ A,
                                                    unsigned mtile, int k0,
                                                    int lane) {
  const unsigned m = mtile * 16u + (lane & 15);
  const int h = lane >> 4;
  const bhalf* row = A + m * (unsigned)LDA;
  const int ka = k0 + 8 * h;
  const int kb = k0 + 8 * h + 16;
  v8bf zz;
#pragma unroll
  for (int i = 0; i < 8; ++i) zz[i] = (bhalf)0.0f;
  union { v16bf v; v8bf p[2]; } u;
  u.p[0] = (ka + 8 <= K) ? *(const v8bf*)(row + ka) : zz;
  u.p[1] = (kb + 8 <= K) ? *(const v8bf*)(row + kb) : zz;
  return u.v;
}

static __device__ __forceinline__ v16bf load_b_packed(const bhalf* __restrict__ Bp,
                                                      unsigned chunk, int lane) {
  return ((const v16bf*)(Bp + chunk * 512u))[lane];
}

// ------------------------- weight packing kernels --------------------------
// Packed layout: chunk (nt,kt) at (nt*KT + kt)*512; element lane*16+i holds
// B[kt*32 + frag_k(lane>>4,i), nt*16 + (lane&15)], zero outside [K)x[N_real).

__global__ void prep_w1(const float* __restrict__ u1, bhalf* __restrict__ out,
                        int P, int N, int KT, int NT) {
  int idx = blockIdx.x * blockDim.x + threadIdx.x;
  if (idx >= NT * KT * 512) return;
  int e = idx & 511, chunk = idx >> 9;
  int kt = chunk % KT, nt = chunk / KT;
  int lane = e >> 4, i = e & 15;
  int k = kt * 32 + frag_k(lane >> 4, i);
  int n = nt * 16 + (lane & 15);
  float v = (k < P) ? u1[k * N + n] : 0.0f;
  out[idx] = (bhalf)v;
}

__global__ void prep_c2(const float* __restrict__ b2w, const float* __restrict__ u2w,
                        bhalf* __restrict__ out, int Q, int H, int KT, int NT) {
  int idx = blockIdx.x * blockDim.x + threadIdx.x;
  if (idx >= NT * KT * 512) return;
  int e = idx & 511, chunk = idx >> 9;
  int kt = chunk % KT, nt = chunk / KT;
  int lane = e >> 4, i = e & 15;
  int k = kt * 32 + frag_k(lane >> 4, i);   // k = q*16 + e
  int n = nt * 16 + (lane & 15);            // n = f*H + h
  int K = Q * 16;
  float v = 0.0f;
  if (k < K) {
    int q = k >> 4, ee = k & 15;
    int f = n / H, hh = n - f * H;
    for (int g = 0; g < 16; ++g)
      v += b2w[ee * 256 + g * 16 + f] * u2w[(q * H + hh) * 16 + g];
  }
  out[idx] = (bhalf)v;
}

__global__ void prep_e(const float* __restrict__ b1w, const float* __restrict__ u3w,
                       bhalf* __restrict__ out, int Rr, int Mo, int KT) {
  int idx = blockIdx.x * blockDim.x + threadIdx.x;
  if (idx >= KT * 512) return;                // NT = 1 (N padded to 16)
  int e = idx & 511, kt = idx >> 9;
  int lane = e >> 4, i = e & 15;
  int k = kt * 32 + frag_k(lane >> 4, i);     // k = f*Rr + r
  int n = lane & 15;                          // n = m
  int K = 16 * Rr;
  float v = 0.0f;
  if (k < K && n < Mo) {
    int f = k / Rr, r = k - f * Rr;
    for (int kk = 0; kk < 16; ++kk)
      v += b1w[f * 16 + kk] * u3w[(r * Mo + n) * 16 + kk];
  }
  out[idx] = (bhalf)v;
}

__global__ void f32_to_bf16(const float* __restrict__ in, bhalf* __restrict__ out,
                            int n) {
  int i = blockIdx.x * blockDim.x + threadIdx.x;
  if (i < n) out[i] = (bhalf)in[i];
}

// -------------------------------- P1 GEMM ----------------------------------
// A = X[z0+z][p][q][r] (k-strided gather, K = P <= 16, one padded k-tile).
// Store T1 permuted to [z][r][d][q][e] so P2 reads plain row-major (K=(q,e)).
template <int P, int Q, int RR, int D>
__global__ __launch_bounds__(128) void gemm_p1(const bhalf* __restrict__ Xin,
                                               const bhalf* __restrict__ W1p,
                                               bhalf* __restrict__ T1, int z0,
                                               int MT) {
  constexpr unsigned QR = Q * RR;
  const int lane = threadIdx.x & 31;
  const int mt = blockIdx.x * 4 + (threadIdx.x >> 5);
  if (mt >= MT) return;
  const int nt = blockIdx.y;
  const int h = lane >> 4;

  v16bf a;
  {
    const unsigned m = mt * 16u + (lane & 15);
    const unsigned z = m / QR, t = m - z * QR;
    const unsigned q = t / RR, r = t - q * RR;
    const unsigned base = ((unsigned)z0 + z) * (P * QR) + q * RR + r;
#pragma unroll
    for (int i = 0; i < 16; ++i) {
      int k = frag_k(h, i);
      a[i] = (k < P) ? Xin[base + (unsigned)k * QR] : (bhalf)0.0f;
    }
  }
  v16bf b = load_b_packed(W1p, nt, lane);  // KT = 1
  v8f acc;
#pragma unroll
  for (int j = 0; j < 8; ++j) acc[j] = 0.0f;
  acc = wmma_bf16(a, b, acc);

  const unsigned nn = nt * 16u + (lane & 15);
  const unsigned d = nn >> 4, ee = nn & 15;
#pragma unroll
  for (int j = 0; j < 8; ++j) {
    unsigned m = mt * 16u + 8u * h + j;
    unsigned z = m / QR, t = m - z * QR;
    unsigned q = t / RR, r = t - q * RR;
    unsigned off = ((z * RR + r) * D + d) * (Q * 16u) + q * 16u + ee;
    T1[off] = (bhalf)acc[j];
  }
}

// -------------------------------- P2 GEMM (dominant) -----------------------
// A = T1 row-major [M2 x K2]; B packed; store T2 permuted to [z][d][h][f][r]
// so P3 reads plain row-major (K=(f,r)). 2x2 WMMA tiles per wave.
template <int Q, int RR, int D, int H>
__global__ __launch_bounds__(128) void gemm_p2(const bhalf* __restrict__ T1,
                                               const bhalf* __restrict__ C2p,
                                               bhalf* __restrict__ T2, int MSUP) {
  constexpr int K = Q * 16;
  constexpr int KT = (K + 31) / 32;
  constexpr unsigned RD = RR * D;
  const int lane = threadIdx.x & 31;
  const int msup = blockIdx.x * 4 + (threadIdx.x >> 5);
  if (msup >= MSUP) return;
  const int nsup = blockIdx.y;
  const unsigned mt0 = msup * 2, nt0 = nsup * 2;

  v8f acc[2][2];
#pragma unroll
  for (int ti = 0; ti < 2; ++ti)
#pragma unroll
    for (int tj = 0; tj < 2; ++tj)
#pragma unroll
      for (int j = 0; j < 8; ++j) acc[ti][tj][j] = 0.0f;

#pragma unroll
  for (int kt = 0; kt < KT; ++kt) {
    const int k0 = kt * 32;
    v16bf a0 = load_a_tile<K, K>(T1, mt0, k0, lane);
    v16bf a1 = load_a_tile<K, K>(T1, mt0 + 1, k0, lane);
    v16bf b0 = load_b_packed(C2p, nt0 * KT + kt, lane);
    v16bf b1 = load_b_packed(C2p, (nt0 + 1) * KT + kt, lane);
    acc[0][0] = wmma_bf16(a0, b0, acc[0][0]);
    acc[0][1] = wmma_bf16(a0, b1, acc[0][1]);
    acc[1][0] = wmma_bf16(a1, b0, acc[1][0]);
    acc[1][1] = wmma_bf16(a1, b1, acc[1][1]);
  }

  const int h = lane >> 4;
  const unsigned nloc = lane & 15;
#pragma unroll
  for (int ti = 0; ti < 2; ++ti) {
#pragma unroll
    for (int tj = 0; tj < 2; ++tj) {
      const unsigned n = (nt0 + tj) * 16u + nloc;
      const unsigned f = n / H, hh = n - f * H;
#pragma unroll
      for (int j = 0; j < 8; ++j) {
        unsigned m = (mt0 + ti) * 16u + 8u * h + j;
        unsigned z = m / RD, t = m - z * RD;
        unsigned r = t / D, d = t - r * D;
        unsigned off = (((z * D + d) * H + hh) * 16u + f) * RR + r;
        T2[off] = (bhalf)acc[ti][tj][j];
      }
    }
  }
}

// -------------------------------- P3 GEMM + ReLU ---------------------------
// A = T2 row-major [M3 x K3]; single n-tile (N padded to 16, real N = MO).
// Output is next block's X: [z0+z][d][h][m] row-major.
template <int RR, int D, int H, int MO>
__global__ __launch_bounds__(128) void gemm_p3(const bhalf* __restrict__ T2,
                                               const bhalf* __restrict__ Ep,
                                               bhalf* __restrict__ Xout, int z0,
                                               int MSUP) {
  constexpr int K = 16 * RR;
  constexpr int KT = K / 32;
  constexpr unsigned DH = D * H;
  const int lane = threadIdx.x & 31;
  const int msup = blockIdx.x * 4 + (threadIdx.x >> 5);
  if (msup >= MSUP) return;
  const unsigned mt0 = msup * 2;

  v8f acc0, acc1;
#pragma unroll
  for (int j = 0; j < 8; ++j) { acc0[j] = 0.0f; acc1[j] = 0.0f; }

#pragma unroll
  for (int kt = 0; kt < KT; ++kt) {
    const int k0 = kt * 32;
    v16bf a0 = load_a_tile<K, K>(T2, mt0, k0, lane);
    v16bf a1 = load_a_tile<K, K>(T2, mt0 + 1, k0, lane);
    v16bf b = load_b_packed(Ep, kt, lane);
    acc0 = wmma_bf16(a0, b, acc0);
    acc1 = wmma_bf16(a1, b, acc1);
  }

  const unsigned n = lane & 15;
  const int h = lane >> 4;
  if (n < MO) {
#pragma unroll
    for (int ti = 0; ti < 2; ++ti) {
#pragma unroll
      for (int j = 0; j < 8; ++j) {
        unsigned m = (mt0 + ti) * 16u + 8u * h + j;
        unsigned z = m / DH, rest = m - z * DH;
        float v = ti ? acc1[j] : acc0[j];
        v = v > 0.0f ? v : 0.0f;
        Xout[((unsigned)z0 + z) * (DH * MO) + rest * MO + n] = (bhalf)v;
      }
    }
  }
}

// Final FC: out[z,j] = b6[j] + sum_c h[z,c] * w6[j,c]   (1.3 MFLOP, VALU f32)
__global__ void fc_kernel(const bhalf* __restrict__ hbuf,
                          const float* __restrict__ w6,
                          const float* __restrict__ b6, float* __restrict__ out,
                          int n) {
  int idx = blockIdx.x * blockDim.x + threadIdx.x;
  if (idx >= n) return;
  int z = idx / 10, j = idx - z * 10;
  float s = b6[j];
  for (int c = 0; c < 256; ++c)
    s += (float)hbuf[z * 256 + c] * w6[j * 256 + c];
  out[idx] = s;
}

// ------------------------------- host side ---------------------------------
template <int P, int Q, int RR, int D, int H, int MO>
static void run_block(void* const* w, const bhalf* Xin, bhalf* Xout, bhalf* W1p,
                      bhalf* C2p, bhalf* Ep, bhalf* T1, bhalf* T2, int Zc,
                      hipStream_t stream) {
  const float* u1 = (const float*)w[0];
  const float* b2 = (const float*)w[1];
  const float* u2 = (const float*)w[2];
  const float* b1 = (const float*)w[3];
  const float* u3 = (const float*)w[4];

  constexpr int N1 = D * 16, NT1 = N1 / 16;
  constexpr int K2 = Q * 16, N2 = 16 * H, KT2 = (K2 + 31) / 32, NT2 = N2 / 16;
  constexpr int K3 = 16 * RR, KT3 = K3 / 32;

  { int n = NT1 * 512;
    prep_w1<<<(n + 255) / 256, 256, 0, stream>>>(u1, W1p, P, N1, 1, NT1); }
  { int n = NT2 * KT2 * 512;
    prep_c2<<<(n + 255) / 256, 256, 0, stream>>>(b2, u2, C2p, Q, H, KT2, NT2); }
  { int n = KT3 * 512;
    prep_e<<<(n + 255) / 256, 256, 0, stream>>>(b1, u3, Ep, RR, MO, KT3); }

  for (int z0 = 0; z0 < 256; z0 += Zc) {
    const int MT1 = Zc * Q * RR / 16;
    gemm_p1<P, Q, RR, D><<<dim3((MT1 + 3) / 4, NT1), 128, 0, stream>>>(
        Xin, W1p, T1, z0, MT1);

    const int MSUP2 = Zc * RR * D / 32;
    gemm_p2<Q, RR, D, H><<<dim3((MSUP2 + 3) / 4, N2 / 32), 128, 0, stream>>>(
        T1, C2p, T2, MSUP2);

    const int MSUP3 = Zc * D * H / 32;
    gemm_p3<RR, D, H, MO><<<dim3((MSUP3 + 3) / 4, 1), 128, 0, stream>>>(
        T2, Ep, Xout, z0, MSUP3);
  }
}

extern "C" void kernel_launch(void* const* d_in, const int* in_sizes, int n_in,
                              void* d_out, int out_size, void* d_ws, size_t ws_size,
                              hipStream_t stream) {
  (void)in_sizes; (void)n_in; (void)out_size;
  const int Z = 256;

  auto aln = [](size_t x) { return (x + 255) & ~(size_t)255; };
  const size_t XBYTES = aln((size_t)Z * 4096 * sizeof(bhalf));
  const size_t W1B = aln(16 * 512 * sizeof(bhalf));
  const size_t C2B = aln(16 * 8 * 512 * sizeof(bhalf));
  const size_t EB  = aln(14 * 512 * sizeof(bhalf));
  const size_t fixedB = 2 * XBYTES + W1B + C2B + EB;

  // Largest z-chunk (multiple of 4) whose T1/T2 scratch fits in ws.
  int Zc = 4;
  const int cands[7] = {256, 128, 64, 32, 16, 8, 4};
  for (int ci = 0; ci < 7; ++ci) {
    size_t need = fixedB + aln((size_t)cands[ci] * 65536 * sizeof(bhalf))
                         + aln((size_t)cands[ci] * 114688 * sizeof(bhalf));
    if (need <= ws_size) { Zc = cands[ci]; break; }
  }

  char* p = (char*)d_ws;
  bhalf* Xping = (bhalf*)p; p += XBYTES;
  bhalf* Xpong = (bhalf*)p; p += XBYTES;
  bhalf* W1p = (bhalf*)p; p += W1B;
  bhalf* C2p = (bhalf*)p; p += C2B;
  bhalf* Ep  = (bhalf*)p; p += EB;
  bhalf* T1  = (bhalf*)p; p += aln((size_t)Zc * 65536 * sizeof(bhalf));
  bhalf* T2  = (bhalf*)p;

  {
    int n = Z * 784;
    f32_to_bf16<<<(n + 255) / 256, 256, 0, stream>>>((const float*)d_in[0], Xping, n);
  }

  // Block 1: (p,q,r)=(4,7,28) -> (16,16,16); blocks 2-4: 16^3 -> 16^3;
  // block 5: 16^3 -> (4,8,8).
  run_block<4, 7, 28, 16, 16, 16>(d_in + 1,  Xping, Xpong, W1p, C2p, Ep, T1, T2, Zc, stream);
  run_block<16, 16, 16, 16, 16, 16>(d_in + 6,  Xpong, Xping, W1p, C2p, Ep, T1, T2, Zc, stream);
  run_block<16, 16, 16, 16, 16, 16>(d_in + 11, Xping, Xpong, W1p, C2p, Ep, T1, T2, Zc, stream);
  run_block<16, 16, 16, 16, 16, 16>(d_in + 16, Xpong, Xping, W1p, C2p, Ep, T1, T2, Zc, stream);
  run_block<16, 16, 16, 4, 8, 8>(d_in + 21, Xping, Xpong, W1p, C2p, Ep, T1, T2, Zc, stream);

  { int n = Z * 10;
    fc_kernel<<<(n + 255) / 256, 256, 0, stream>>>(
        Xpong, (const float*)d_in[26], (const float*)d_in[27], (float*)d_out, n); }
}
